// PyTorchCrossAttention_10187662426849
// MI455X (gfx1250) — compile-verified, matrix operand order checked by reading, not run on hardware
//
#include <hip/hip_runtime.h>
#include <cstdint>

// ---------------------------------------------------------------------------
// MI455X (gfx1250) cross-attention pipeline:
//   cvt f32->f16  ->  gemm_wmma(Q,+bias,+rmsnorm) -> gemm_wmma(K) -> gemm_wmma(V, transposed out)
//   -> attn_wmma (flash, online softmax)  ->  gemm_wmma(O, f32 out)
// All matmuls use v_wmma_f32_16x16x32_f16; all tile staging uses the Tensor
// Data Mover (tensor_load_to_lds, 6-arg clang-23 form) with TENSORcnt waits
// and double buffering.
// ---------------------------------------------------------------------------

typedef __attribute__((ext_vector_type(16))) _Float16 v16h;
typedef __attribute__((ext_vector_type(8)))  _Float16 v8h;
typedef __attribute__((ext_vector_type(4)))  _Float16 v4h;
typedef __attribute__((ext_vector_type(8)))  float    v8f;
typedef __attribute__((ext_vector_type(4)))  float    v4f;
typedef __attribute__((ext_vector_type(4)))  unsigned int u32x4;
typedef __attribute__((ext_vector_type(8)))  int      i32x8;
typedef __attribute__((ext_vector_type(4)))  int      i32x4;

#define RMS_EPS 1.1920928955078125e-07f   // torch.finfo(float32).eps

// Generic->LDS byte offset: flat shared addresses carry the LDS offset in [31:0].
__device__ __forceinline__ unsigned lds_off(const void* p) {
  return (unsigned)(uintptr_t)p;
}

// 2D TDM tile load: tile_rows x tile_cols f16 elements, row stride in elements.
// LDS padding enabled: pad_amt DWORD-code inserted every 2^(pad_int+1) DWORDs.
// Descriptor packing per CDNA5 ISA §8.3 (group0) / §8.4 (group1).
__device__ __forceinline__ void tdm_load_2d(unsigned lds_byte, const void* gptr,
                                            unsigned tile_cols, unsigned tile_rows,
                                            unsigned long long row_stride_elts,
                                            unsigned pad_int_code, unsigned pad_amt_code) {
  unsigned long long ga = (unsigned long long)(uintptr_t)gptr;
  u32x4 g0;
  g0[0] = 1u;                                            // count=1, user descriptor
  g0[1] = lds_byte;                                      // lds_addr
  g0[2] = (unsigned)ga;                                  // global_addr[31:0]
  g0[3] = ((unsigned)(ga >> 32) & 0x1FFFFFFu) | (2u << 30); // addr[56:32] | type=2
  unsigned td0 = tile_cols, td1 = tile_rows;
  i32x8 g1;
  g1[0] = (int)((1u << 16)                       // data_size = 2 bytes (f16)
              | (1u << 20)                       // pad_enable
              | (pad_int_code << 22)
              | (pad_amt_code << 25));
  g1[1] = (int)((td0 & 0xFFFFu) << 16);                          // tensor_dim0 lo
  g1[2] = (int)(((td0 >> 16) & 0xFFFFu) | ((td1 & 0xFFFFu) << 16)); // dim0 hi | dim1 lo
  g1[3] = (int)(((td1 >> 16) & 0xFFFFu) | ((tile_cols & 0xFFFFu) << 16)); // dim1 hi | tile_dim0
  g1[4] = (int)(tile_rows & 0xFFFFu);                            // tile_dim1 (tile_dim2=0)
  g1[5] = (int)(unsigned)(row_stride_elts & 0xFFFFFFFFull);      // dim0_stride lo
  g1[6] = (int)(unsigned)((row_stride_elts >> 32) & 0xFFFFull);  // dim0_stride hi
  g1[7] = 0;
  i32x4 gz4 = (i32x4)0;                                  // groups 2/3 unused (2D)
  i32x8 gz8 = (i32x8)0;
  __builtin_amdgcn_tensor_load_to_lds(g0, g1, gz4, gz4, gz8, 0);
}

// Load a 16x32 f16 WMMA A/B fragment from an LDS tile row (padded row stride).
// ISA layout: half-wave 0 -> K {kk..kk+7, kk+16..kk+23}, half 1 -> {kk+8.., kk+24..}.
__device__ __forceinline__ v16h frag_ld(const char* rowbase, int kk, int half) {
  const v8h* p0 = (const v8h*)(rowbase + (kk + half * 8) * 2);
  const v8h* p1 = (const v8h*)(rowbase + (kk + 16 + half * 8) * 2);
  v8h lo = *p0, hi = *p1;
  v16h r;
#pragma unroll
  for (int i = 0; i < 8; ++i) { r[i] = lo[i]; r[i + 8] = hi[i]; }
  return r;
}

__device__ __forceinline__ v8f wmma_f16(v16h a, v16h b, v8f c) {
  return __builtin_amdgcn_wmma_f32_16x16x32_f16(false, a, false, b, (short)0, c,
                                                false, false);
}

// ---------------------------------------------------------------------------
// fp32 -> fp16 conversion (vectorized)
// ---------------------------------------------------------------------------
__global__ void cvt_f16_kernel(const float* __restrict__ in,
                               _Float16* __restrict__ out, long n) {
  long i = ((long)blockIdx.x * blockDim.x + threadIdx.x) * 4;
  if (i + 3 < n) {
    v4f x = *(const v4f*)(in + i);
    v4h h;
#pragma unroll
    for (int j = 0; j < 4; ++j) h[j] = (_Float16)x[j];
    *(v4h*)(out + i) = h;
  }
}

// ---------------------------------------------------------------------------
// Tiled WMMA GEMM:  C[M,N] = A[M,K] * W[N,K]^T (+bias) (+per-128-col RMSNorm)
// Block tile 128x128, BK=64, 256 threads = 8 waves (4M x 2N), wave tile 32x64.
// TDM double-buffered LDS staging; 16B pad per 128B row (interval code 4,
// amount code 3) to spread LDS banks.
// out_mode 0: f32 row-major [M,N]
// out_mode 1: f16 [b, head, s, d]   (head = blockIdx.x, d = col-in-head)
// out_mode 2: f16 [b, head, d, s]   (transposed, for attention V)
// LDS map (byte offsets from smem):
//   A buf0: 0        A buf1: 18432
//   B buf0: 36864    B buf1: 55296
//   red[]:  73728    (total 74240)
// ---------------------------------------------------------------------------
__global__ void gemm_wmma(const _Float16* __restrict__ A,
                          const _Float16* __restrict__ W,
                          const float* __restrict__ bias,
                          const float* __restrict__ normw,
                          void* __restrict__ out,
                          int M, int N, int K, int out_mode, int Srows, int NH) {
  extern __shared__ char smem[];
  const int ROWB  = 64 * 2 + 16;               // 144B padded row
  const int TILEB = 128 * ROWB;                // 18432B per tile buffer

  const int tid  = threadIdx.x;
  const int lane = tid & 31, half = lane >> 4, ln = lane & 15;
  const int wave = tid >> 5, wm = wave >> 1, wn = wave & 1;
  const long m0 = (long)blockIdx.y * 128;
  const long n0 = (long)blockIdx.x * 128;
  const int KT = K / 64;

  v8f acc[2][4];
#pragma unroll
  for (int mt = 0; mt < 2; ++mt)
#pragma unroll
    for (int nt = 0; nt < 4; ++nt) acc[mt][nt] = (v8f)0.0f;

  if (tid < 32) {                               // wave 0 drives the TDM
    tdm_load_2d(lds_off(smem + 0 * TILEB), A + m0 * K, 64, 128, (unsigned)K, 4, 3);
    tdm_load_2d(lds_off(smem + 2 * TILEB), W + n0 * K, 64, 128, (unsigned)K, 4, 3);
    __builtin_amdgcn_s_wait_tensorcnt(0);
  }
  __syncthreads();

  int cur = 0;
  for (int kt = 0; kt < KT; ++kt) {
    if (tid < 32 && kt + 1 < KT) {              // prefetch next tiles (async)
      tdm_load_2d(lds_off(smem + (cur ^ 1) * TILEB),
                  A + m0 * K + (long)(kt + 1) * 64, 64, 128, (unsigned)K, 4, 3);
      tdm_load_2d(lds_off(smem + (2 + (cur ^ 1)) * TILEB),
                  W + n0 * K + (long)(kt + 1) * 64, 64, 128, (unsigned)K, 4, 3);
    }
    const char* Acur = smem + cur * TILEB;
    const char* Bcur = smem + (2 + cur) * TILEB;
#pragma unroll
    for (int ks = 0; ks < 2; ++ks) {
      const int kk = ks * 32;
      v16h a0 = frag_ld(Acur + (wm * 32 + ln) * ROWB, kk, half);
      v16h a1 = frag_ld(Acur + (wm * 32 + 16 + ln) * ROWB, kk, half);
#pragma unroll
      for (int nt = 0; nt < 4; ++nt) {
        v16h b = frag_ld(Bcur + (wn * 64 + nt * 16 + ln) * ROWB, kk, half);
        acc[0][nt] = wmma_f16(a0, b, acc[0][nt]);
        acc[1][nt] = wmma_f16(a1, b, acc[1][nt]);
      }
    }
    if (tid < 32) __builtin_amdgcn_s_wait_tensorcnt(0);
    __syncthreads();
    cur ^= 1;
  }

  // ---- epilogue: bias ----
  if (bias) {
#pragma unroll
    for (int nt = 0; nt < 4; ++nt) {
      float bv = bias[n0 + wn * 64 + nt * 16 + ln];
#pragma unroll
      for (int mt = 0; mt < 2; ++mt)
#pragma unroll
        for (int r = 0; r < 8; ++r) acc[mt][nt][r] += bv;
    }
  }

  // ---- epilogue: per-row RMSNorm over the 128 block columns (== one head) ----
  float rs[2][8];
  float nwv[4];
#pragma unroll
  for (int nt = 0; nt < 4; ++nt) nwv[nt] = 1.0f;
#pragma unroll
  for (int mt = 0; mt < 2; ++mt)
#pragma unroll
    for (int r = 0; r < 8; ++r) rs[mt][r] = 1.0f;

  if (normw) {
    float* red = (float*)(smem + 4 * TILEB);
    if (tid < 128) red[tid] = 0.0f;
    __syncthreads();
#pragma unroll
    for (int mt = 0; mt < 2; ++mt)
#pragma unroll
      for (int r = 0; r < 8; ++r) {
        float s = 0.0f;
#pragma unroll
        for (int nt = 0; nt < 4; ++nt) s += acc[mt][nt][r] * acc[mt][nt][r];
        s += __shfl_xor(s, 1, 32);
        s += __shfl_xor(s, 2, 32);
        s += __shfl_xor(s, 4, 32);
        s += __shfl_xor(s, 8, 32);               // sum over this wave's 64 cols
        if (ln == 0)
          atomicAdd(&red[wm * 32 + mt * 16 + half * 8 + r], s);  // ds_add_f32
      }
    __syncthreads();
#pragma unroll
    for (int mt = 0; mt < 2; ++mt)
#pragma unroll
      for (int r = 0; r < 8; ++r)
        rs[mt][r] = rsqrtf(red[wm * 32 + mt * 16 + half * 8 + r] * (1.0f / 128.0f)
                           + RMS_EPS);
#pragma unroll
    for (int nt = 0; nt < 4; ++nt) nwv[nt] = normw[wn * 64 + nt * 16 + ln];
  }

  // ---- epilogue: store ----
#pragma unroll
  for (int mt = 0; mt < 2; ++mt)
#pragma unroll
    for (int nt = 0; nt < 4; ++nt)
#pragma unroll
      for (int r = 0; r < 8; ++r) {
        long row = m0 + wm * 32 + mt * 16 + half * 8 + r;
        int  ci  = wn * 64 + nt * 16 + ln;       // col within 128-wide block/head
        float v  = acc[mt][nt][r] * rs[mt][r] * nwv[nt];
        if (out_mode == 0) {
          ((float*)out)[row * (long)N + n0 + ci] = v;
        } else {
          long bb = row / Srows, s = row % Srows;
          long idx = (out_mode == 1)
                   ? (((bb * NH + blockIdx.x) * (long)Srows + s) * 128 + ci)
                   : (((bb * NH + blockIdx.x) * 128L + ci) * Srows + s);
          ((_Float16*)out)[idx] = (_Float16)v;
        }
      }
}

// ---------------------------------------------------------------------------
// Flash attention, GQA (28 q-heads / 4 kv-heads, G=7), D=128.
// Block = 128 q rows of one head; 8 waves, each wave owns 16 rows x full
// 128-wide KV tile -> softmax row stats are wave-local (shfl over the 16-lane
// half matching the C-fragment row layout). K and V^T tiles TDM double-
// buffered; P converted C->A layout through a wave-private LDS patch
// (same-wave DS ordering, no barrier needed).
// LDS map (tile = 34816B): Q @0, K @1+buf, V @3+buf, P @5 (8 waves x 16 x 272B)
// ---------------------------------------------------------------------------
__global__ void attn_wmma(const _Float16* __restrict__ qh,
                          const _Float16* __restrict__ kh,
                          const _Float16* __restrict__ vth,
                          _Float16* __restrict__ ah) {
  extern __shared__ char smem[];
  const int ROWB = 128 * 2 + 16;                 // 272B padded row
  const int TILEB = 128 * ROWB;                  // 34816B

  const int tid  = threadIdx.x;
  const int lane = tid & 31, half = lane >> 4, ln = lane & 15;
  const int w    = tid >> 5;
  const int mb = blockIdx.x, h = blockIdx.y, b = blockIdx.z;
  const int kv = h / 7;
  const int S = 2048, SKV = 1024, D = 128, NH = 28, NKV = 4;

  const _Float16* Qg = qh + (((long)b * NH + h) * S + (long)mb * 128) * D;
  const _Float16* Kg = kh + (((long)b * NKV + kv) * SKV) * D;
  const _Float16* Vg = vth + (((long)b * NKV + kv) * D) * (long)SKV; // [d][skv]

  if (tid < 32) {
    tdm_load_2d(lds_off(smem + 0 * TILEB), Qg, 128, 128, 128, 5, 3);
    tdm_load_2d(lds_off(smem + 1 * TILEB), Kg, 128, 128, 128, 5, 3);
    tdm_load_2d(lds_off(smem + 3 * TILEB), Vg, 128, 128, SKV, 5, 3);
    __builtin_amdgcn_s_wait_tensorcnt(0);
  }
  __syncthreads();

  v16h qa[4];
#pragma unroll
  for (int ks = 0; ks < 4; ++ks)
    qa[ks] = frag_ld(smem + (w * 16 + ln) * ROWB, ks * 32, half);

  float mst[8], lst[8];
  v8f oacc[8];
#pragma unroll
  for (int r = 0; r < 8; ++r) { mst[r] = -1e30f; lst[r] = 0.0f; }
#pragma unroll
  for (int dt = 0; dt < 8; ++dt) oacc[dt] = (v8f)0.0f;

  const float sc = 0.088388347648318447f;        // 1/sqrt(128)
  int cur = 0;
  for (int t = 0; t < 8; ++t) {                  // Skv / 128 tiles
    if (tid < 32 && t + 1 < 8) {
      tdm_load_2d(lds_off(smem + (1 + (cur ^ 1)) * TILEB),
                  Kg + (long)(t + 1) * 128 * D, 128, 128, 128, 5, 3);
      tdm_load_2d(lds_off(smem + (3 + (cur ^ 1)) * TILEB),
                  Vg + (long)(t + 1) * 128, 128, 128, SKV, 5, 3);
    }
    const char* Kcur = smem + (1 + cur) * TILEB;
    const char* Vcur = smem + (3 + cur) * TILEB;

    // ---- S = Q K^T (16 x 128 per wave) ----
    v8f sacc[8];
#pragma unroll
    for (int nt = 0; nt < 8; ++nt) sacc[nt] = (v8f)0.0f;
#pragma unroll
    for (int nt = 0; nt < 8; ++nt)
#pragma unroll
      for (int ks = 0; ks < 4; ++ks) {
        v16h kb = frag_ld(Kcur + (nt * 16 + ln) * ROWB, ks * 32, half);
        sacc[nt] = wmma_f16(qa[ks], kb, sacc[nt]);
      }

    // ---- online softmax (rows shared across the 16-lane half) ----
#pragma unroll
    for (int r = 0; r < 8; ++r) {
      float mx = mst[r];
#pragma unroll
      for (int nt = 0; nt < 8; ++nt) mx = fmaxf(mx, sacc[nt][r] * sc);
      mx = fmaxf(mx, __shfl_xor(mx, 1, 32));
      mx = fmaxf(mx, __shfl_xor(mx, 2, 32));
      mx = fmaxf(mx, __shfl_xor(mx, 4, 32));
      mx = fmaxf(mx, __shfl_xor(mx, 8, 32));
      float alpha = __expf(mst[r] - mx);
      float rsum = 0.0f;
#pragma unroll
      for (int nt = 0; nt < 8; ++nt) {
        float p = __expf(sacc[nt][r] * sc - mx);
        sacc[nt][r] = p;
        rsum += p;
      }
      rsum += __shfl_xor(rsum, 1, 32);
      rsum += __shfl_xor(rsum, 2, 32);
      rsum += __shfl_xor(rsum, 4, 32);
      rsum += __shfl_xor(rsum, 8, 32);
      lst[r] = lst[r] * alpha + rsum;
      mst[r] = mx;
#pragma unroll
      for (int dt = 0; dt < 8; ++dt) oacc[dt][r] *= alpha;
    }

    // ---- P: C-layout -> A-layout through wave-private LDS ----
    char* Pw = smem + 5 * TILEB + w * (16 * ROWB);
#pragma unroll
    for (int nt = 0; nt < 8; ++nt)
#pragma unroll
      for (int r = 0; r < 8; ++r)
        *(_Float16*)(Pw + (half * 8 + r) * ROWB + (nt * 16 + ln) * 2) =
            (_Float16)sacc[nt][r];

    v16h pa[4];
#pragma unroll
    for (int ks = 0; ks < 4; ++ks)
      pa[ks] = frag_ld(Pw + ln * ROWB, ks * 32, half);

    // ---- O += P V  (V^T tile rows are d, contiguous in kv) ----
#pragma unroll
    for (int dt = 0; dt < 8; ++dt)
#pragma unroll
      for (int ks = 0; ks < 4; ++ks) {
        v16h vb = frag_ld(Vcur + (dt * 16 + ln) * ROWB, ks * 32, half);
        oacc[dt] = wmma_f16(pa[ks], vb, oacc[dt]);
      }

    if (tid < 32) __builtin_amdgcn_s_wait_tensorcnt(0);
    __syncthreads();
    cur ^= 1;
  }

  // ---- epilogue: normalize and store attn as [b, s, h, d] f16 ----
  float linv[8];
#pragma unroll
  for (int r = 0; r < 8; ++r) linv[r] = 1.0f / lst[r];
#pragma unroll
  for (int dt = 0; dt < 8; ++dt)
#pragma unroll
    for (int r = 0; r < 8; ++r) {
      long s = (long)mb * 128 + w * 16 + half * 8 + r;
      int  d = dt * 16 + ln;
      ah[(((long)b * S + s) * NH + h) * 128 + d] =
          (_Float16)(oacc[dt][r] * linv[r]);
    }
}

// ---------------------------------------------------------------------------
// Host side
// ---------------------------------------------------------------------------
extern "C" void kernel_launch(void* const* d_in, const int* in_sizes, int n_in,
                              void* d_out, int out_size, void* d_ws, size_t ws_size,
                              hipStream_t stream) {
  const float* hs  = (const float*)d_in[0];   // [2,2048,3584]
  const float* mm  = (const float*)d_in[1];   // [2,1024,3584]
  const float* q_w = (const float*)d_in[2];   // [3584,3584]
  const float* q_b = (const float*)d_in[3];
  const float* k_w = (const float*)d_in[4];   // [512,3584]
  const float* k_b = (const float*)d_in[5];
  const float* v_w = (const float*)d_in[6];
  const float* v_b = (const float*)d_in[7];
  const float* o_w = (const float*)d_in[8];   // [3584,3584]
  const float* qn  = (const float*)d_in[9];
  const float* kn  = (const float*)d_in[10];
  float* out = (float*)d_out;

  const long nHS = 2L * 2048 * 3584;   // 14,680,064
  const long nMM = 2L * 1024 * 3584;   //  7,340,032
  const long nQW = 3584L * 3584;       // 12,845,056
  const long nKW = 512L * 3584;        //  1,835,008
  const long nQH = 2L * 28 * 2048 * 128;
  const long nKH = 2L * 4 * 1024 * 128;

  char* ws = (char*)d_ws;               // ~158 MB total scratch
  _Float16* hh  = (_Float16*)ws;  ws += nHS * 2;
  _Float16* mmh = (_Float16*)ws;  ws += nMM * 2;
  _Float16* qwh = (_Float16*)ws;  ws += nQW * 2;
  _Float16* kwh = (_Float16*)ws;  ws += nKW * 2;
  _Float16* vwh = (_Float16*)ws;  ws += nKW * 2;
  _Float16* owh = (_Float16*)ws;  ws += nQW * 2;
  _Float16* qhb = (_Float16*)ws;  ws += nQH * 2;  // [b,h,s,d]
  _Float16* khb = (_Float16*)ws;  ws += nKH * 2;  // [b,kv,skv,d]
  _Float16* vtb = (_Float16*)ws;  ws += nKH * 2;  // [b,kv,d,skv]
  _Float16* ahb = (_Float16*)ws;  ws += nQH * 2;  // [b,s,h*d]

  // fp32 -> fp16 staging
  auto cvt = [&](const float* src, _Float16* dst, long n) {
    long blocks = (n / 4 + 255) / 256;
    cvt_f16_kernel<<<dim3((unsigned)blocks), dim3(256), 0, stream>>>(src, dst, n);
  };
  cvt(hs, hh, nHS);
  cvt(mm, mmh, nMM);
  cvt(q_w, qwh, nQW);
  cvt(k_w, kwh, nKW);
  cvt(v_w, vwh, nKW);
  cvt(o_w, owh, nQW);

  const size_t gemm_lds = (size_t)(4 * 128 * 144) + 512;   // 74,240 B
  const size_t attn_lds = (size_t)(6 * 128 * 272);         // 208,896 B

  // Q = rmsnorm(hs @ q_w^T + q_b)  -> [b,h,s,d] f16
  gemm_wmma<<<dim3(28, 32), dim3(256), gemm_lds, stream>>>(
      hh, qwh, q_b, qn, qhb, 4096, 3584, 3584, 1, 2048, 28);
  // K = rmsnorm(mm @ k_w^T + k_b)  -> [b,kv,skv,d] f16
  gemm_wmma<<<dim3(4, 16), dim3(256), gemm_lds, stream>>>(
      mmh, kwh, k_b, kn, khb, 2048, 512, 3584, 1, 1024, 4);
  // V = mm @ v_w^T + v_b           -> [b,kv,d,skv] f16 (transposed for PV B-frags)
  gemm_wmma<<<dim3(4, 16), dim3(256), gemm_lds, stream>>>(
      mmh, vwh, v_b, nullptr, vtb, 2048, 512, 3584, 2, 1024, 4);

  // attention: grid (S/128, heads, batch)
  attn_wmma<<<dim3(16, 28, 2), dim3(256), attn_lds, stream>>>(qhb, khb, vtb, ahb);

  // out = attn @ o_w^T  (f32)
  gemm_wmma<<<dim3(28, 32), dim3(256), gemm_lds, stream>>>(
      ahb, owh, nullptr, nullptr, out, 4096, 3584, 3584, 0, 2048, 28);
}
